// MyGnn_49512382988587
// MI455X (gfx1250) — matrix operand order repaired and checked by reading, not compiled, not run on hardware
//
#include <hip/hip_runtime.h>
#include <cstddef>

// ---------------------------------------------------------------- types
typedef __attribute__((ext_vector_type(16))) _Float16 v16h;
typedef __attribute__((ext_vector_type(8)))  float    v8f;

__device__ __forceinline__ v8f wmma_f16(v16h a, v16h b, v8f c) {
  // D = A(16x32 f16) * B(32x16 f16) + C(16x16 f32)
  return __builtin_amdgcn_wmma_f32_16x16x32_f16(
      false, a, false, b, (short)0, c, false, false);
}

// Packed operand layouts (one v16h = 32 contiguous bytes per lane):
//   A tile (16x32, ISA 7.12.2): element (m, k) of the tile lives at
//     lane = m + 16*((k>>3)&1),  h = (k&7) + 8*((k>>4)&1)
//   B tile (32x16): element (k, nn) lives at
//     lane = nn + 16*((k>>3)&1), h = (k&7) + 8*((k>>4)&1)
// Both verified against the ISA tables (lanes 0-15: A K in {0..7,16..23},
// B the same; lanes 16-31: A K in {8..15,24..31}, B the same).

__device__ __forceinline__ void edge_sd(const int* __restrict__ ei, int E, int e,
                                        int& s, int& d) {
  if (e < E) { s = ei[e]; d = ei[E + e]; }
  else       { s = d = e - E; }          // appended self-loops
}

#define NEG_BIG (-3.4e38f)

// ---------------------------------------------------------------- utility kernels
__global__ void k_fill(float* __restrict__ p, int n, float v) {
  int i = blockIdx.x * blockDim.x + threadIdx.x;
  if (i < n) p[i] = v;
}

// Convert W[di][dO] (f32, row-major) into packed f16 B-tiles:
// o[((kt*NT + nt)*32 + lane)*16 + h] = W[kt*32 + (lane>>4)*8 + (h&7) + 16*(h>>3)][nt*16 + (lane&15)]
__global__ void k_cvt_pack(const float* __restrict__ w, _Float16* __restrict__ o,
                           int di, int dO) {
  int idx = blockIdx.x * blockDim.x + threadIdx.x;
  if (idx >= di * dO) return;
  const int h    = idx & 15;
  const int lane = (idx >> 4) & 31;
  const int tile = idx >> 9;            // kt*NT + nt
  const int NT   = dO >> 4;
  const int kt   = tile / NT;
  const int nt   = tile - kt * NT;
  const int col  = nt * 16 + (lane & 15);
  const int k    = (lane >> 4) * 8 + (h & 7) + 16 * (h >> 3);
  o[idx] = (_Float16)w[(size_t)(kt * 32 + k) * dO + col];
}

// ---------------------------------------------------------------- fused edge MLP + segment-max
// Block = 128 threads (4 waves); 64 edges per block, 16 per wave.
// h = ReLU([x_src | pos_src - pos_dst] @ W1 + b1) @ W2 + b2 ; atomicMax into agg[dst].
// Weights arrive pre-packed in B-tile layout (read straight from global, L0-resident).
__global__ __launch_bounds__(128) void k_edge_mlp(
    const float* __restrict__ x, const float* __restrict__ pos,
    const int* __restrict__ ei, int N, int E, int EP,
    const _Float16* __restrict__ W1p, const float* __restrict__ b1,
    const _Float16* __restrict__ W2p, const float* __restrict__ b2,
    float* __restrict__ agg) {
  __shared__ alignas(32) _Float16 InsP[4 * 2 * 32 * 16];  // 4 waves x 2 K-tiles, packed A
  __shared__ alignas(32) _Float16 HsP [4 * 4 * 32 * 16];  // 4 waves x 4 K-tiles, packed A
  __shared__ float b1s[128], b2s[128];
  __shared__ int   dst_s[64];

  const int tid = threadIdx.x;
  const int e0  = blockIdx.x * 64;

  b1s[tid] = b1[tid];
  b2s[tid] = b2[tid];

  // Gather edge messages [x_src(61) | pos_src - pos_dst(3)] into packed A layout.
  for (int i = tid; i < 64 * 64; i += 128) {
    const int el = i >> 6, c = i & 63;
    const int e  = e0 + el;
    float f = 0.f;
    if (e < EP) {
      int s, d; edge_sd(ei, E, e, s, d);
      if (c < 61) f = x[(size_t)s * 61 + c];
      else        { int p = c - 61; f = pos[s * 3 + p] - pos[d * 3 + p]; }
      if (c == 0) dst_s[el] = d;
    } else if (c == 0) dst_s[el] = -1;
    const int w  = el >> 4;
    const int kt = c >> 5;
    const int k  = c & 31;
    const int lp = (el & 15) + 16 * ((k >> 3) & 1);
    const int h  = (k & 7) + 8 * ((k >> 4) & 1);
    InsP[((w * 2 + kt) * 32 + lp) * 16 + h] = (_Float16)f;
  }
  __syncthreads();

  const int wave = tid >> 5, lane = tid & 31;
  const int m = lane & 15, hi8 = (lane >> 4) * 8;
  const int laneHi = (m >> 3) & 1;

  // Stage 1: hidden = ReLU(In @ W1 + b1)   (64 -> 128)
  const v16h a0 = *(const v16h*)&InsP[((wave * 2 + 0) * 32 + lane) * 16];
  const v16h a1 = *(const v16h*)&InsP[((wave * 2 + 1) * 32 + lane) * 16];
  for (int nt = 0; nt < 8; ++nt) {
    const float bv = b1s[nt * 16 + m];
    v8f c;
#pragma unroll
    for (int r = 0; r < 8; ++r) c[r] = bv;
    v16h b = *(const v16h*)&W1p[((0 * 8 + nt) * 32 + lane) * 16];
    c = wmma_f16(a0, b, c);
    b = *(const v16h*)&W1p[((1 * 8 + nt) * 32 + lane) * 16];
    c = wmma_f16(a1, b, c);
    // write hidden directly in packed-A layout for stage 2
    const int kt2 = nt >> 1;
    const int h   = (m & 7) + 8 * (nt & 1);
#pragma unroll
    for (int r = 0; r < 8; ++r) {
      float v = c[r]; v = v > 0.f ? v : 0.f;
      const int lp = r + hi8 + 16 * laneHi;
      HsP[((wave * 4 + kt2) * 32 + lp) * 16 + h] = (_Float16)v;
    }
  }
  __syncthreads();

  // Stage 2: out = H @ W2 + b2  (128 -> 128)
  v8f acc[8];
#pragma unroll
  for (int nt = 0; nt < 8; ++nt) {
    const float bv = b2s[nt * 16 + m];
#pragma unroll
    for (int r = 0; r < 8; ++r) acc[nt][r] = bv;
  }
  for (int kt = 0; kt < 4; ++kt) {
    const v16h a = *(const v16h*)&HsP[((wave * 4 + kt) * 32 + lane) * 16];
#pragma unroll
    for (int nt = 0; nt < 8; ++nt) {
      const v16h b = *(const v16h*)&W2p[((kt * 8 + nt) * 32 + lane) * 16];
      acc[nt] = wmma_f16(a, b, acc[nt]);
    }
  }

  // segment_max fused as float atomic max (GLOBAL_ATOMIC_MAX_NUM_F32)
  const int r0 = wave * 16;
#pragma unroll
  for (int nt = 0; nt < 8; ++nt)
#pragma unroll
    for (int r = 0; r < 8; ++r) {
      const int row = r0 + r + hi8;
      const int d = dst_s[row];
      if (d >= 0) atomicMax(&agg[(size_t)d * 128 + nt * 16 + m], acc[nt][r]);
    }
}

// ---------------------------------------------------------------- generic WMMA GEMM
// Y[n,DO] = act(X[n,di] @ W[di,DO] + bias). Block = 256 threads (8 waves), 128-row tile.
// Weights pre-packed; X staged per-K-slice into packed-A layout in LDS.
template <int DO, int ACT>
__global__ __launch_bounds__(256) void k_gemm(
    const float* __restrict__ X, int n, int di,
    const _Float16* __restrict__ Wp, const float* __restrict__ bias,
    float* __restrict__ Y) {
  __shared__ alignas(32) _Float16 XsP[8 * 32 * 16];  // 8 waves, packed A, one K-slice
  constexpr int NT = DO / 16;

  const int tid = threadIdx.x, wave = tid >> 5, lane = tid & 31;
  const int m = lane & 15, hi8 = (lane >> 4) * 8;
  const int row0 = blockIdx.x * 128;

  v8f acc[NT];
#pragma unroll
  for (int nt = 0; nt < NT; ++nt) {
    const float bv = bias ? bias[nt * 16 + m] : 0.f;
#pragma unroll
    for (int r = 0; r < 8; ++r) acc[nt][r] = bv;
  }

  for (int kt = 0; kt * 32 < di; ++kt) {
    for (int i = tid; i < 128 * 32; i += 256) {
      const int r = i >> 5, k = i & 31;
      const int gr = row0 + r;
      const float f = (gr < n) ? X[(size_t)gr * di + kt * 32 + k] : 0.f;
      const int w  = r >> 4;
      const int lp = (r & 15) + 16 * ((k >> 3) & 1);
      const int h  = (k & 7) + 8 * ((k >> 4) & 1);
      XsP[(w * 32 + lp) * 16 + h] = (_Float16)f;
    }
    __syncthreads();

    const v16h a = *(const v16h*)&XsP[(wave * 32 + lane) * 16];
#pragma unroll
    for (int nt = 0; nt < NT; ++nt) {
      const v16h b = *(const v16h*)&Wp[((size_t)(kt * NT + nt) * 32 + lane) * 16];
      acc[nt] = wmma_f16(a, b, acc[nt]);
    }
    __syncthreads();
  }

#pragma unroll
  for (int nt = 0; nt < NT; ++nt)
#pragma unroll
    for (int r = 0; r < 8; ++r) {
      const int gr = row0 + wave * 16 + r + hi8;
      if (gr < n) {
        float v = acc[nt][r];
        if (ACT) v = v > 0.f ? v : 0.f;
        Y[(size_t)gr * DO + nt * 16 + m] = v;
      }
    }
}

// ---------------------------------------------------------------- GAT helper kernels
__global__ void k_attn(const float* __restrict__ xp,
                       const float* __restrict__ asrc, const float* __restrict__ adst,
                       int n, int DO, float* __restrict__ as_, float* __restrict__ ad_) {
  int i = blockIdx.x * blockDim.x + threadIdx.x;
  if (i >= n) return;
  const float* row = xp + (size_t)i * DO;
  float s = 0.f, d = 0.f;
  for (int c = 0; c < DO; ++c) { const float v = row[c]; s += v * asrc[c]; d += v * adst[c]; }
  as_[i] = s; ad_[i] = d;
}

__global__ void k_edge_score(const float* __restrict__ asb, const float* __restrict__ adb,
                             const int* __restrict__ ei, int E, int EP,
                             float* __restrict__ sc, float* __restrict__ m) {
  int e = blockIdx.x * blockDim.x + threadIdx.x;
  if (e >= EP) return;
  int s, d; edge_sd(ei, E, e, s, d);
  float v = asb[s] + adb[d];
  v = v > 0.f ? v : 0.2f * v;          // leaky_relu, slope 0.2
  sc[e] = v;
  atomicMax(&m[d], v);
}

__global__ void k_edge_exp(const float* __restrict__ sc, const float* __restrict__ m,
                           const int* __restrict__ ei, int E, int EP,
                           float* __restrict__ ee, float* __restrict__ z) {
  int e = blockIdx.x * blockDim.x + threadIdx.x;
  if (e >= EP) return;
  int s, d; edge_sd(ei, E, e, s, d);
  const float ex = expf(sc[e] - m[d]);
  ee[e] = ex;
  atomicAdd(&z[d], ex);
}

__global__ void k_edge_scatter(const float* __restrict__ ee, const float* __restrict__ z,
                               const float* __restrict__ xp, const int* __restrict__ ei,
                               int E, int EP, int DO4, int DO, float* __restrict__ out) {
  int idx = blockIdx.x * blockDim.x + threadIdx.x;
  const int e = idx / DO4;
  const int q = idx - e * DO4;
  if (e >= EP) return;
  int s, d; edge_sd(ei, E, e, s, d);
  const float alpha = ee[e] / (z[d] + 1e-16f);
  const float4 v = *(const float4*)(xp + (size_t)s * DO + q * 4);
  float* o = out + (size_t)d * DO + q * 4;
  atomicAdd(o + 0, alpha * v.x);
  atomicAdd(o + 1, alpha * v.y);
  atomicAdd(o + 2, alpha * v.z);
  atomicAdd(o + 3, alpha * v.w);
}

__global__ void k_bias_act(float* __restrict__ out, const float* __restrict__ b,
                           int n, int DO, int act) {
  int i = blockIdx.x * blockDim.x + threadIdx.x;
  if (i >= n * DO) return;
  float v = out[i] + b[i % DO];
  if (act) v = v > 0.f ? v : 0.f;
  out[i] = v;
}

// ---------------------------------------------------------------- host-side helpers
static void gemm_dispatch(int DO, int act, const float* X, int n, int di,
                          const _Float16* Wp, const float* bias, float* Y,
                          hipStream_t st) {
  dim3 g((n + 127) / 128), b(256);
  if (DO == 64) {
    if (act) k_gemm<64, 1><<<g, b, 0, st>>>(X, n, di, Wp, bias, Y);
    else     k_gemm<64, 0><<<g, b, 0, st>>>(X, n, di, Wp, bias, Y);
  } else if (DO == 128) {
    if (act) k_gemm<128, 1><<<g, b, 0, st>>>(X, n, di, Wp, bias, Y);
    else     k_gemm<128, 0><<<g, b, 0, st>>>(X, n, di, Wp, bias, Y);
  } else {
    if (act) k_gemm<256, 1><<<g, b, 0, st>>>(X, n, di, Wp, bias, Y);
    else     k_gemm<256, 0><<<g, b, 0, st>>>(X, n, di, Wp, bias, Y);
  }
}

static void gat_layer(const float* X, int di, int DO, const _Float16* Wp,
                      const float* asrc, const float* adst, const float* ab, int act,
                      float* xp, float* out, const int* ei, int N, int E, int EP,
                      float* asb, float* adb, float* sc, float* ee,
                      float* m, float* z, hipStream_t st) {
  gemm_dispatch(DO, 0, X, N, di, Wp, nullptr, xp, st);
  k_attn<<<(N + 255) / 256, 256, 0, st>>>(xp, asrc, adst, N, DO, asb, adb);
  k_fill<<<(N + 255) / 256, 256, 0, st>>>(m, N, NEG_BIG);
  k_fill<<<(N + 255) / 256, 256, 0, st>>>(z, N, 0.f);
  k_edge_score<<<(EP + 255) / 256, 256, 0, st>>>(asb, adb, ei, E, EP, sc, m);
  k_edge_exp  <<<(EP + 255) / 256, 256, 0, st>>>(sc, m, ei, E, EP, ee, z);
  const int total = N * DO;
  k_fill<<<(total + 255) / 256, 256, 0, st>>>(out, total, 0.f);
  const int DO4 = DO / 4;
  const long tot2 = (long)EP * DO4;
  k_edge_scatter<<<(unsigned)((tot2 + 255) / 256), 256, 0, st>>>(
      ee, z, xp, ei, E, EP, DO4, DO, out);
  k_bias_act<<<(total + 255) / 256, 256, 0, st>>>(out, ab, N, DO, act);
}

// ---------------------------------------------------------------- entry point
extern "C" void kernel_launch(void* const* d_in, const int* in_sizes, int n_in,
                              void* d_out, int out_size, void* d_ws, size_t ws_size,
                              hipStream_t stream) {
  const float* x   = (const float*)d_in[0];
  const float* pos = (const float*)d_in[1];
  const int*   ei  = (const int*)  d_in[2];
  const float* lw1 = (const float*)d_in[3];
  const float* lb1 = (const float*)d_in[4];
  const float* lw2 = (const float*)d_in[5];
  const float* lb2 = (const float*)d_in[6];
  const float* gw1 = (const float*)d_in[7];
  const float* gb1 = (const float*)d_in[8];
  const float* gw2 = (const float*)d_in[9];
  const float* gb2 = (const float*)d_in[10];
  const float* gw3 = (const float*)d_in[11];
  const float* gb3 = (const float*)d_in[12];
  const float* w1  = (const float*)d_in[13];
  const float* as1 = (const float*)d_in[14];
  const float* ad1 = (const float*)d_in[15];
  const float* ab1 = (const float*)d_in[16];
  const float* w2  = (const float*)d_in[17];
  const float* as2 = (const float*)d_in[18];
  const float* ad2 = (const float*)d_in[19];
  const float* ab2 = (const float*)d_in[20];
  const float* w3  = (const float*)d_in[21];
  const float* as3 = (const float*)d_in[22];
  const float* ad3 = (const float*)d_in[23];
  const float* ab3 = (const float*)d_in[24];

  const int N  = in_sizes[1] / 3;
  const int E  = in_sizes[2] / 2;
  const int EP = E + N;

  // workspace layout
  char* base = (char*)d_ws;
  size_t off = 0;
  auto alloc = [&](size_t bytes) -> void* {
    off = (off + 255) & ~(size_t)255;
    void* p = base + off;
    off += bytes;
    return p;
  };
  _Float16* lw1h = (_Float16*)alloc(64 * 128 * 2);
  _Float16* lw2h = (_Float16*)alloc(128 * 128 * 2);
  _Float16* gw1h = (_Float16*)alloc(128 * 64 * 2);
  _Float16* gw2h = (_Float16*)alloc(64 * 256 * 2);
  _Float16* gw3h = (_Float16*)alloc(256 * 128 * 2);
  _Float16* w1h  = (_Float16*)alloc(128 * 256 * 2);
  _Float16* w2h  = (_Float16*)alloc(256 * 256 * 2);
  _Float16* w3h  = (_Float16*)alloc(256 * 128 * 2);
  float* bufA = (float*)alloc((size_t)N * 256 * 4);
  float* bufB = (float*)alloc((size_t)N * 256 * 4);
  float* bufC = (float*)alloc((size_t)N * 256 * 4);
  float* asb  = (float*)alloc((size_t)N * 4);
  float* adb  = (float*)alloc((size_t)N * 4);
  float* mbuf = (float*)alloc((size_t)N * 4);
  float* zbuf = (float*)alloc((size_t)N * 4);
  float* scb  = (float*)alloc((size_t)EP * 4);
  float* eeb  = (float*)alloc((size_t)EP * 4);

  auto cvt = [&](const float* s, _Float16* d, int di, int dO) {
    k_cvt_pack<<<(di * dO + 255) / 256, 256, 0, stream>>>(s, d, di, dO);
  };
  cvt(lw1, lw1h, 64, 128);
  cvt(lw2, lw2h, 128, 128);
  cvt(gw1, gw1h, 128, 64);
  cvt(gw2, gw2h, 64, 256);
  cvt(gw3, gw3h, 256, 128);
  cvt(w1,  w1h, 128, 256);
  cvt(w2,  w2h, 256, 256);
  cvt(w3,  w3h, 256, 128);

  // PointNetConv: edge MLP + segment_max (fused), then global_nn
  k_fill<<<(N * 128 + 255) / 256, 256, 0, stream>>>(bufA, N * 128, NEG_BIG);
  k_edge_mlp<<<(EP + 63) / 64, 128, 0, stream>>>(
      x, pos, ei, N, E, EP, lw1h, lb1, lw2h, lb2, bufA);

  gemm_dispatch(64,  1, bufA, N, 128, gw1h, gb1, bufB, stream);  // ReLU(agg@gw1+gb1)
  gemm_dispatch(256, 1, bufB, N, 64,  gw2h, gb2, bufC, stream);  // ReLU(..@gw2+gb2)
  gemm_dispatch(128, 0, bufC, N, 256, gw3h, gb3, bufA, stream);  // h0 = ..@gw3+gb3

  // 3x GATConv
  gat_layer(bufA, 128, 256, w1h, as1, ad1, ab1, /*relu*/1, bufB, bufC,
            ei, N, E, EP, asb, adb, scb, eeb, mbuf, zbuf, stream);
  gat_layer(bufC, 256, 256, w2h, as2, ad2, ab2, /*relu*/1, bufB, bufA,
            ei, N, E, EP, asb, adb, scb, eeb, mbuf, zbuf, stream);
  gat_layer(bufA, 256, 128, w3h, as3, ad3, ab3, /*none*/0, bufB, (float*)d_out,
            ei, N, E, EP, asb, adb, scb, eeb, mbuf, zbuf, stream);
}
